// CausalAttention_56873956934253
// MI455X (gfx1250) — matrix operand order repaired and checked by reading, not compiled, not run on hardware
//
#include <hip/hip_runtime.h>
#include <hip/hip_bf16.h>

typedef __attribute__((ext_vector_type(16))) __bf16 v16bf;
typedef __attribute__((ext_vector_type(8)))  __bf16 v8bf;
typedef __attribute__((ext_vector_type(8)))  float  v8f;

#define S_LEN  1024
#define C_OUT  512
#define IC     384
#define NH     8
#define HD     64

__device__ __forceinline__ unsigned short f2bf(float x) {
  unsigned int u = __builtin_bit_cast(unsigned int, x);
  u += 0x7FFFu + ((u >> 16) & 1u);          // round-to-nearest-even
  return (unsigned short)(u >> 16);
}
__device__ __forceinline__ v8bf ld8(const unsigned short* p) {
  return *(const v8bf*)p;                   // 16B LDS/global load (aligned by construction)
}
__device__ __forceinline__ v16bf cat8(v8bf a, v8bf b) {
  return __builtin_shufflevector(a, b, 0,1,2,3,4,5,6,7,8,9,10,11,12,13,14,15);
}
// CDNA5 async global->LDS b128 copy (ASYNCcnt-tracked, bypasses VGPRs)
__device__ __forceinline__ void async_b128(unsigned lds_byte_addr, const void* gaddr) {
  asm volatile("global_load_async_to_lds_b128 %0, %1, off"
               :: "v"(lds_byte_addr), "v"(gaddr) : "memory");
}
__device__ __forceinline__ void wait_async0() {
  asm volatile("s_wait_asynccnt 0x0" ::: "memory");
}

// ---------------- Kernel 1: weight-norm -> bf16 weights [3][C][IC] ----------
__global__ __launch_bounds__(128)
void wn_kernel(const float* __restrict__ Wq_v, const float* __restrict__ Wq_g,
               const float* __restrict__ Wk_v, const float* __restrict__ Wk_g,
               const float* __restrict__ Wv_v, const float* __restrict__ Wv_g,
               unsigned short* __restrict__ wn) {
  const int row = blockIdx.x;           // 0..511
  const int mat = blockIdx.y;           // 0..2
  const float* v = (mat == 0) ? Wq_v : (mat == 1) ? Wk_v : Wv_v;
  const float* g = (mat == 0) ? Wq_g : (mat == 1) ? Wk_g : Wv_g;
  const float* vr = v + (size_t)row * IC;
  float ss = 0.f;
  for (int i = threadIdx.x; i < IC; i += 128) { float x = vr[i]; ss += x * x; }
  #pragma unroll
  for (int m = 16; m >= 1; m >>= 1) ss += __shfl_xor(ss, m, 32);
  __shared__ float red[4];
  if ((threadIdx.x & 31) == 0) red[threadIdx.x >> 5] = ss;
  __syncthreads();
  float scale = g[row] * rsqrtf(red[0] + red[1] + red[2] + red[3]);
  unsigned short* o = wn + ((size_t)mat * C_OUT + row) * IC;
  for (int i = threadIdx.x; i < IC; i += 128) o[i] = f2bf(vr[i] * scale);
}

// ---------------- Kernel 2: projection Y[b][h][s][d] = X(s,ic) @ W^T + bias --
__global__ __launch_bounds__(256)
void proj_kernel(const float* __restrict__ X,           // [B][IC][S]
                 const unsigned short* __restrict__ W,  // [C][IC] bf16
                 const float* __restrict__ bias,        // [C]
                 unsigned short* __restrict__ Y,        // [B][NH][S][HD] bf16
                 float scale) {
  const int st = blockIdx.x, ot = blockIdx.y, b = blockIdx.z;
  const int sbase = st * 32, ocbase = ot * 64;
  __shared__ __align__(16) unsigned short Xs[32][40];
  __shared__ __align__(16) unsigned short Ws[64][40];
  const int tid = threadIdx.x;
  const int lane = tid & 31, wave = tid >> 5;
  const int mt = wave >> 2, nt = wave & 3;
  const int half = lane >> 4, l16 = lane & 15;
  v8f acc = {};
  for (int k0 = 0; k0 < IC; k0 += 32) {
    // W tile 64x32 bf16: pure byte-move -> async global->LDS (1 b128/thread)
    {
      int oc = tid >> 1, hrow = tid & 1;                  // 256 threads = 128 rows x 2 halves
      async_b128((unsigned)(size_t)&Ws[oc][hrow * 16],
                 W + (size_t)(ocbase + oc) * IC + k0 + hrow * 16);
    }
    // X tile 32x32: fp32 -> bf16 convert (must go through VGPRs)
    #pragma unroll
    for (int j = 0; j < 4; ++j) {
      int i = tid + 256 * j;
      int ic = i >> 5, s = i & 31;
      Xs[s][ic] = f2bf(X[((size_t)b * IC + k0 + ic) * S_LEN + sbase + s]);
    }
    wait_async0();
    __syncthreads();
    const unsigned short* ar = &Xs[mt * 16 + l16][0];
    v16bf A  = cat8(ld8(ar + half * 8), ld8(ar + 16 + half * 8));
    const unsigned short* br = &Ws[nt * 16 + l16][0];
    v16bf Bv = cat8(ld8(br + half * 8), ld8(br + 16 + half * 8));
    acc = __builtin_amdgcn_wmma_f32_16x16x32_bf16(false, A, false, Bv,
                                                  (short)0, acc, false, false);
    __syncthreads();
  }
  const int col = ocbase + nt * 16 + l16;
  const float bcol = bias[col];
  const int h = col >> 6, d = col & 63;
  #pragma unroll
  for (int r = 0; r < 8; ++r) {
    int s = sbase + mt * 16 + half * 8 + r;
    Y[(((size_t)b * NH + h) * S_LEN + s) * HD + d] = f2bf((acc[r] + bcol) * scale);
  }
}

// ---------------- Kernel 3: causal flash attention ---------------------------
__global__ __launch_bounds__(128)
void attn_kernel(const unsigned short* __restrict__ Qh,  // [B][NH][S][HD] bf16, pre-scaled
                 const unsigned short* __restrict__ Kh,
                 const unsigned short* __restrict__ Vh,
                 float* __restrict__ out) {               // [B][C][S] fp32
  const int qt = blockIdx.x;                // 16 q tiles of 64
  const int bh = blockIdx.y;                // 64 (b,h)
  const int b = bh >> 3, h = bh & 7;
  const int qbase = qt * 64;
  const int tid = threadIdx.x;
  const int lane = tid & 31, wq = tid >> 5; // 4 waves, 16 q-rows each
  const int half = lane >> 4, l16 = lane & 15;

  __shared__ __align__(16) unsigned short Ks[32][72];     // K tile row-major
  __shared__ __align__(16) unsigned short Vt[64][40];     // V tile transposed [d][kv]
  __shared__ __align__(16) unsigned short Ps[4][16][40];  // per-wave P bounce

  const size_t bhS = ((size_t)b * NH + h) * S_LEN;

  // Q A-fragments (K = 64 split into 2x32)
  const unsigned short* qrow = Qh + (bhS + qbase + wq * 16 + l16) * HD;
  v16bf aq0 = cat8(ld8(qrow +      half * 8), ld8(qrow + 16 + half * 8));
  v16bf aq1 = cat8(ld8(qrow + 32 + half * 8), ld8(qrow + 48 + half * 8));

  v8f o0 = {}, o1 = {}, o2 = {}, o3 = {};
  float mrun[8], lrun[8];
  #pragma unroll
  for (int r = 0; r < 8; ++r) { mrun[r] = -1e30f; lrun[r] = 0.f; }

  const int nkt = 2 * (qt + 1);             // causal: only kv < qbase+64
  for (int kt = 0; kt < nkt; ++kt) {
    const int kvbase = kt * 32;
    {   // K tile 32x64 bf16: async global->LDS, 2 b128/thread
      int row = tid >> 2, chunk = tid & 3;
      const unsigned short* src = Kh + (bhS + kvbase + row) * HD + chunk * 16;
      unsigned dst = (unsigned)(size_t)&Ks[row][chunk * 16];
      async_b128(dst,      src);
      async_b128(dst + 16, src + 8);
    }
    {   // V tile transposed (layout change -> manual)
      int kv = tid >> 2, chunk = tid & 3;
      const unsigned short* src = Vh + (bhS + kvbase + kv) * HD + chunk * 16;
      #pragma unroll
      for (int e = 0; e < 16; ++e) Vt[chunk * 16 + e][kv] = src[e];
    }
    wait_async0();
    __syncthreads();

    // S = Q K^T : two 16x16 tiles over the 32 kv columns
    v8f s0 = {}, s1 = {};
    {
      const unsigned short* kr0 = &Ks[l16][0];
      const unsigned short* kr1 = &Ks[16 + l16][0];
      v16bf b00 = cat8(ld8(kr0 +      half * 8), ld8(kr0 + 16 + half * 8));
      v16bf b01 = cat8(ld8(kr0 + 32 + half * 8), ld8(kr0 + 48 + half * 8));
      v16bf b10 = cat8(ld8(kr1 +      half * 8), ld8(kr1 + 16 + half * 8));
      v16bf b11 = cat8(ld8(kr1 + 32 + half * 8), ld8(kr1 + 48 + half * 8));
      s0 = __builtin_amdgcn_wmma_f32_16x16x32_bf16(false, aq0, false, b00, (short)0, s0, false, false);
      s0 = __builtin_amdgcn_wmma_f32_16x16x32_bf16(false, aq1, false, b01, (short)0, s0, false, false);
      s1 = __builtin_amdgcn_wmma_f32_16x16x32_bf16(false, aq0, false, b10, (short)0, s1, false, false);
      s1 = __builtin_amdgcn_wmma_f32_16x16x32_bf16(false, aq1, false, b11, (short)0, s1, false, false);
    }

    // strict causal mask + online softmax (row m = half*8+r lives in 16-lane half)
    float f[8];
    #pragma unroll
    for (int r = 0; r < 8; ++r) {
      const int q_idx = qbase + wq * 16 + half * 8 + r;
      float v0 = ((kvbase      + l16) < q_idx) ? s0[r] : -10000.f;
      float v1 = ((kvbase + 16 + l16) < q_idx) ? s1[r] : -10000.f;
      float mx = fmaxf(v0, v1);
      mx = fmaxf(mx, __shfl_xor(mx, 1, 32));
      mx = fmaxf(mx, __shfl_xor(mx, 2, 32));
      mx = fmaxf(mx, __shfl_xor(mx, 4, 32));
      mx = fmaxf(mx, __shfl_xor(mx, 8, 32));
      float newm = fmaxf(mrun[r], mx);
      float p0 = __expf(v0 - newm);
      float p1 = __expf(v1 - newm);
      float ps = p0 + p1;
      ps += __shfl_xor(ps, 1, 32);
      ps += __shfl_xor(ps, 2, 32);
      ps += __shfl_xor(ps, 4, 32);
      ps += __shfl_xor(ps, 8, 32);
      float fr = __expf(mrun[r] - newm);
      lrun[r] = lrun[r] * fr + ps;
      mrun[r] = newm;
      f[r] = fr;
      Ps[wq][half * 8 + r][l16]      = f2bf(p0);
      Ps[wq][half * 8 + r][16 + l16] = f2bf(p1);
    }
    #pragma unroll
    for (int r = 0; r < 8; ++r) { o0[r] *= f[r]; o1[r] *= f[r]; o2[r] *= f[r]; o3[r] *= f[r]; }

    // O += P @ V
    {
      const unsigned short* pr = &Ps[wq][l16][0];
      v16bf ap = cat8(ld8(pr + half * 8), ld8(pr + 16 + half * 8));
      const unsigned short* v0p = &Vt[ 0 + l16][0];
      const unsigned short* v1p = &Vt[16 + l16][0];
      const unsigned short* v2p = &Vt[32 + l16][0];
      const unsigned short* v3p = &Vt[48 + l16][0];
      v16bf bv0 = cat8(ld8(v0p + half * 8), ld8(v0p + 16 + half * 8));
      v16bf bv1 = cat8(ld8(v1p + half * 8), ld8(v1p + 16 + half * 8));
      v16bf bv2 = cat8(ld8(v2p + half * 8), ld8(v2p + 16 + half * 8));
      v16bf bv3 = cat8(ld8(v3p + half * 8), ld8(v3p + 16 + half * 8));
      o0 = __builtin_amdgcn_wmma_f32_16x16x32_bf16(false, ap, false, bv0, (short)0, o0, false, false);
      o1 = __builtin_amdgcn_wmma_f32_16x16x32_bf16(false, ap, false, bv1, (short)0, o1, false, false);
      o2 = __builtin_amdgcn_wmma_f32_16x16x32_bf16(false, ap, false, bv2, (short)0, o2, false, false);
      o3 = __builtin_amdgcn_wmma_f32_16x16x32_bf16(false, ap, false, bv3, (short)0, o3, false, false);
    }
    __syncthreads();
  }

  // normalize, zero row 0 (start_mask), write out [B][C][H][W] = [B][C][S]
  #pragma unroll
  for (int r = 0; r < 8; ++r) {
    const int q_idx = qbase + wq * 16 + half * 8 + r;
    const float inv = (q_idx == 0) ? 0.f : 1.f / lrun[r];
    const size_t base = ((size_t)b * C_OUT + h * HD) * S_LEN + q_idx;
    out[base + (size_t)( 0 + l16) * S_LEN] = o0[r] * inv;
    out[base + (size_t)(16 + l16) * S_LEN] = o1[r] * inv;
    out[base + (size_t)(32 + l16) * S_LEN] = o2[r] * inv;
    out[base + (size_t)(48 + l16) * S_LEN] = o3[r] * inv;
  }
}

extern "C" void kernel_launch(void* const* d_in, const int* in_sizes, int n_in,
                              void* d_out, int out_size, void* d_ws, size_t ws_size,
                              hipStream_t stream) {
  (void)in_sizes; (void)n_in; (void)out_size; (void)ws_size;
  const float* q    = (const float*)d_in[0];
  const float* k    = (const float*)d_in[1];
  const float* Wq_v = (const float*)d_in[2];
  const float* Wq_g = (const float*)d_in[3];
  const float* bq   = (const float*)d_in[4];
  const float* Wk_v = (const float*)d_in[5];
  const float* Wk_g = (const float*)d_in[6];
  const float* bk   = (const float*)d_in[7];
  const float* Wv_v = (const float*)d_in[8];
  const float* Wv_g = (const float*)d_in[9];
  const float* bv   = (const float*)d_in[10];
  float* out = (float*)d_out;

  char* ws = (char*)d_ws;
  const size_t szW  = (size_t)C_OUT * IC * 2;          // one bf16 weight matrix
  const size_t szP  = (size_t)8 * NH * S_LEN * HD * 2; // one bf16 projection
  unsigned short* Wn = (unsigned short*)ws;            // [3][C][IC]
  unsigned short* Qh = (unsigned short*)(ws + 3 * szW);
  unsigned short* Kh = (unsigned short*)(ws + 3 * szW + szP);
  unsigned short* Vh = (unsigned short*)(ws + 3 * szW + 2 * szP);

  wn_kernel<<<dim3(C_OUT, 3), 128, 0, stream>>>(Wq_v, Wq_g, Wk_v, Wk_g, Wv_v, Wv_g, Wn);
  proj_kernel<<<dim3(32, 8, 8), 256, 0, stream>>>(q, Wn,                  bq, Qh, 0.125f);
  proj_kernel<<<dim3(32, 8, 8), 256, 0, stream>>>(k, Wn + C_OUT * IC,     bk, Kh, 1.0f);
  proj_kernel<<<dim3(32, 8, 8), 256, 0, stream>>>(k, Wn + 2 * C_OUT * IC, bv, Vh, 1.0f);
  attn_kernel<<<dim3(16, 64), 128, 0, stream>>>(Qh, Kh, Vh, out);
}
